// Attention_58351425683518
// MI455X (gfx1250) — matrix operand order repaired and checked by reading, not compiled, not run on hardware
//
#include <hip/hip_runtime.h>
#include <hip/hip_bf16.h>
#include <math.h>

typedef __attribute__((ext_vector_type(16))) __bf16 v16bf;
typedef __attribute__((ext_vector_type(8)))  __bf16 v8bf;
typedef __attribute__((ext_vector_type(8)))  float  v8f;

// CDNA5 has a hardware V_TANH_F32 transcendental; use it when clang exposes it.
#if __has_builtin(__builtin_amdgcn_tanhf)
__device__ __forceinline__ float fast_tanh(float x) { return __builtin_amdgcn_tanhf(x); }
#elif __has_builtin(__builtin_amdgcn_tanh_f32)
__device__ __forceinline__ float fast_tanh(float x) { return __builtin_amdgcn_tanh_f32(x); }
#else
__device__ __forceinline__ float fast_tanh(float x) { return tanhf(x); }
#endif

// ---------------- conversion kernels (f32 -> bf16 staging) ----------------

__global__ void cvt_plain(__bf16* __restrict__ dst, const float* __restrict__ src, int n) {
    int i = blockIdx.x * blockDim.x + threadIdx.x;
    if (i < n) dst[i] = (__bf16)src[i];
}

// hidden (1024 x 512) -> hB, and also into hcat[:, 512:1024]
__global__ void cvt_hidden(__bf16* __restrict__ hB, __bf16* __restrict__ hcat,
                           const float* __restrict__ hidden, int n) {
    int i = blockIdx.x * blockDim.x + threadIdx.x;
    if (i < n) {
        __bf16 v = (__bf16)hidden[i];
        hB[i] = v;
        int row = i >> 9;        // /512
        int col = i & 511;
        hcat[(size_t)row * 1024 + 512 + col] = v;
    }
}

// split W_attn (512 x 1024) into W_h (512x512) and W_e (512x512), bf16
__global__ void cvt_wsplit(__bf16* __restrict__ WhB, __bf16* __restrict__ WeB,
                           const float* __restrict__ W_attn, int n) {
    int i = blockIdx.x * blockDim.x + threadIdx.x;
    if (i < n) {
        int row = i >> 10;       // /1024
        int col = i & 1023;
        __bf16 v = (__bf16)W_attn[i];
        if (col < 512) WhB[(size_t)row * 512 + col] = v;
        else           WeB[(size_t)row * 512 + (col - 512)] = v;
    }
}

// ---------------- WMMA GEMM: C[M,N] = A[M,K](bf16) * W[N,K](bf16)^T ----------------
// One wave computes one 16x16 tile. Optional bias[n] add, optional tanh epilogue.

template<bool ADD_BIAS, bool DO_TANH>
__global__ void wmma_gemm_nt(const __bf16* __restrict__ A,
                             const __bf16* __restrict__ W,
                             const float*  __restrict__ bias,
                             float* __restrict__ C,
                             int M, int N, int K) {
    int wave = blockIdx.x * (blockDim.x >> 5) + (threadIdx.x >> 5);
    int lane = threadIdx.x & 31;
    int tilesN = N >> 4;
    int totalTiles = (M >> 4) * tilesN;
    if (wave >= totalTiles) return;           // wave-uniform: EXEC stays all-1s

    int tm = (wave / tilesN) << 4;
    int tn = (wave % tilesN) << 4;
    int half = lane >> 4;                      // 0: lanes 0-15, 1: lanes 16-31
    int l16  = lane & 15;

    // A fragment: lane holds row (tm+l16); 16 contiguous K at offset 16*half
    const __bf16* aRow = A + (size_t)(tm + l16) * K + (half << 4);
    // B fragment: lane holds col (tn+l16); 16 contiguous K at offset 16*half
    const __bf16* bRow = W + (size_t)(tn + l16) * K + (half << 4);

    v8f acc = {};
    for (int k0 = 0; k0 < K; k0 += 32) {
        v8bf alo = *(const v8bf*)(aRow + k0);
        v8bf ahi = *(const v8bf*)(aRow + k0 + 8);
        v16bf av;
#pragma unroll
        for (int i = 0; i < 8; ++i) { av[i] = alo[i]; av[i + 8] = ahi[i]; }
        v16bf bv = *(const v16bf*)(bRow + k0);
        acc = __builtin_amdgcn_wmma_f32_16x16x32_bf16(
                  false, av, false, bv, (short)0, acc, false, false);
    }

    int cn = tn + l16;
    float badd = ADD_BIAS ? bias[cn] : 0.0f;
#pragma unroll
    for (int r = 0; r < 8; ++r) {
        int cm = tm + r + (half << 3);         // C VGPR r -> row r (+8 for upper half)
        float v = acc[r] + badd;
        if (DO_TANH) v = fast_tanh(v);
        C[(size_t)cm * N + cn] = v;
    }
}

// ---------------- fused energies + masked softmax + weighted context ----------------
// One block per (b,t). ep already contains +b_attn bias.
// Energies phase: one wave per s, lanes split k with float4 loads (coalesced),
// then a 5-step shuffle reduction across the wave32.

__global__ void attn_fused(const float* __restrict__ hp,
                           const float* __restrict__ ep,
                           const float* __restrict__ enc,
                           const float* __restrict__ mask,
                           const float* __restrict__ Wv,
                           const float* __restrict__ bv,
                           float* __restrict__ attn_out,
                           float* __restrict__ wc_out,
                           __bf16* __restrict__ hcat) {
    const int S = 512, D = 512, T = 128, E = 512;
    int bt = blockIdx.x;
    int b  = bt / T;
    int tid  = threadIdx.x;                    // 256 threads = 8 waves
    int lane = tid & 31;
    int wv32 = tid >> 5;

    __shared__ alignas(16) float sh_hp[512];
    __shared__ alignas(16) float sh_wv[512];
    __shared__ float sh_e[512];
    __shared__ float sh_m[512];
    __shared__ float red[256];

    for (int k = tid; k < D; k += 256) {
        sh_hp[k] = hp[(size_t)bt * D + k];
        sh_wv[k] = Wv[k];
    }
    __syncthreads();

    float bvv = bv[0];
    // ---- energies: wave w handles s = w, w+8, ..., lanes split k ----
    for (int s = wv32; s < S; s += 8) {
        const float* er = ep + ((size_t)b * S + s) * D;
        float acc = 0.0f;
#pragma unroll
        for (int k = lane * 4; k < D; k += 128) {
            float4 ev = *(const float4*)(er + k);
            float4 hv = *(const float4*)(sh_hp + k);
            float4 wv = *(const float4*)(sh_wv + k);
            acc += fast_tanh(hv.x + ev.x) * wv.x;
            acc += fast_tanh(hv.y + ev.y) * wv.y;
            acc += fast_tanh(hv.z + ev.z) * wv.z;
            acc += fast_tanh(hv.w + ev.w) * wv.w;
        }
#pragma unroll
        for (int off = 16; off > 0; off >>= 1)
            acc += __shfl_xor(acc, off, 32);
        if (lane == 0) {
            float m = mask[b * S + s];
            sh_m[s] = m;
            // energies = (acc+bv)*m, then softmax input x = energies*m
            sh_e[s] = (acc + bvv) * m * m;
        }
    }
    __syncthreads();

    // ---- block max ----
    float lmax = -3.402823466e+38f;
    for (int s = tid; s < S; s += 256) lmax = fmaxf(lmax, sh_e[s]);
    red[tid] = lmax; __syncthreads();
    for (int o = 128; o > 0; o >>= 1) {
        if (tid < o) red[tid] = fmaxf(red[tid], red[tid + o]);
        __syncthreads();
    }
    float mx = red[0];
    __syncthreads();

    // ---- exp * mask, block sum ----
    float lsum = 0.0f;
    for (int s = tid; s < S; s += 256) {
        float e = __expf(sh_e[s] - mx) * sh_m[s];
        sh_e[s] = e;
        lsum += e;
    }
    red[tid] = lsum; __syncthreads();
    for (int o = 128; o > 0; o >>= 1) {
        if (tid < o) red[tid] += red[tid + o];
        __syncthreads();
    }
    float inv = 1.0f / (red[0] + 1e-6f);
    __syncthreads();

    for (int s = tid; s < S; s += 256) {
        float a = sh_e[s] * inv;
        sh_e[s] = a;
        attn_out[(size_t)bt * S + s] = a;
    }
    __syncthreads();

    // ---- weighted context: wc[e] = sum_s attn[s] * enc[b,s,e] ----
    for (int e = tid; e < E; e += 256) {
        const float* ec = enc + (size_t)b * S * E + e;
        float acc = 0.0f;
#pragma unroll 8
        for (int s = 0; s < S; ++s) acc += sh_e[s] * ec[(size_t)s * E];
        wc_out[(size_t)bt * E + e] = acc;
        hcat[(size_t)bt * 1024 + e] = (__bf16)acc;   // left half of h_cat
    }
}

// ---------------- host-side launcher ----------------

extern "C" void kernel_launch(void* const* d_in, const int* in_sizes, int n_in,
                              void* d_out, int out_size, void* d_ws, size_t ws_size,
                              hipStream_t stream) {
    const int B = 8, T = 128, S = 512, E = 512, D = 512;
    const int BT = B * T;       // 1024
    const int BS = B * S;       // 4096

    const float* hidden = (const float*)d_in[0];
    const float* enc    = (const float*)d_in[1];
    const float* mask   = (const float*)d_in[2];
    const float* W_attn = (const float*)d_in[3];
    const float* b_attn = (const float*)d_in[4];
    const float* W_v    = (const float*)d_in[5];
    const float* b_v    = (const float*)d_in[6];
    const float* W_out  = (const float*)d_in[7];

    float* out_htilde = (float*)d_out;                   // B*T*D
    float* out_wc     = out_htilde + (size_t)BT * D;     // B*T*E
    float* out_attn   = out_wc     + (size_t)BT * E;     // B*T*S

    // workspace layout (all 256B-aligned by construction)
    char* p = (char*)d_ws;
    float*  hp   = (float*)p;            p += (size_t)BT * D * 4;   // 2 MB
    float*  ep   = (float*)p;            p += (size_t)BS * D * 4;   // 8 MB
    __bf16* hB   = (__bf16*)p;           p += (size_t)BT * D * 2;   // 1 MB
    __bf16* eB   = (__bf16*)p;           p += (size_t)BS * E * 2;   // 4 MB
    __bf16* WhB  = (__bf16*)p;           p += (size_t)D * D * 2;    // 0.5 MB
    __bf16* WeB  = (__bf16*)p;           p += (size_t)D * E * 2;    // 0.5 MB
    __bf16* WoB  = (__bf16*)p;           p += (size_t)D * (E + D) * 2; // 1 MB
    __bf16* hcat = (__bf16*)p;           p += (size_t)BT * (E + D) * 2; // 2 MB

    // 1. stage bf16 copies
    {
        int n = BS * E;   // 2,097,152
        cvt_plain<<<(n + 255) / 256, 256, 0, stream>>>(eB, enc, n);
    }
    {
        int n = D * (E + D); // 524,288
        cvt_plain<<<(n + 255) / 256, 256, 0, stream>>>(WoB, W_out, n);
    }
    {
        int n = BT * D;   // 524,288
        cvt_hidden<<<(n + 255) / 256, 256, 0, stream>>>(hB, hcat, hidden, n);
    }
    {
        int n = D * (E + D); // 524,288
        cvt_wsplit<<<(n + 255) / 256, 256, 0, stream>>>(WhB, WeB, W_attn, n);
    }

    // 2. hp = hidden @ W_h^T   (M=1024, N=512, K=512)
    {
        int tiles = (BT / 16) * (D / 16);   // 2048 waves
        wmma_gemm_nt<false, false><<<(tiles + 7) / 8, 256, 0, stream>>>(
            hB, WhB, nullptr, hp, BT, D, D);
    }
    // 3. ep = enc @ W_e^T + b_attn   (M=4096, N=512, K=512)
    {
        int tiles = (BS / 16) * (D / 16);   // 8192 waves
        wmma_gemm_nt<true, false><<<(tiles + 7) / 8, 256, 0, stream>>>(
            eB, WeB, b_attn, ep, BS, D, E);
    }

    // 4. fused energies / softmax / weighted-context (writes attn, wc, hcat left half)
    attn_fused<<<BT, 256, 0, stream>>>(hp, ep, enc, mask, W_v, b_v,
                                       out_attn, out_wc, hcat);

    // 5. h_tilde = tanh(h_cat @ W_out^T)   (M=1024, N=512, K=1024)
    {
        int tiles = (BT / 16) * (D / 16);   // 2048 waves
        wmma_gemm_nt<false, true><<<(tiles + 7) / 8, 256, 0, stream>>>(
            hcat, WoB, nullptr, out_htilde, BT, D, E + D);
    }
}
